// ClusterContrastLoss_446676599051
// MI455X (gfx1250) — matrix operand order repaired and checked by reading, not compiled, not run on hardware
//
#include <hip/hip_runtime.h>
#include <hip/hip_bf16.h>
#include <math.h>

// ---------------------------------------------------------------------------
// ClusterContrastLoss for MI455X (gfx1250, wave32).
//   * logits GEMMs on v_wmma_f32_16x16x32_f16 (unit-norm inputs -> f16-safe)
//   * tile fills via GLOBAL_LOAD_ASYNC_TO_LDS_B128 (ASYNCcnt), no VGPR bounce
//   * B tiles double-buffered: next tile's DMA issued before waiting on the
//     current one; s_wait_asynccnt 12 (partial, in-order completion) overlaps
//     DMA with WMMA + exp/log VALU work
//   * LDS tiles padded to 392 halves/row (196 dwords % 64 banks = 4) so the
//     16-lane fragment loads stripe across all 64 banks (no conflicts)
//   * kt-outer / ns-inner WMMA loop: 1 A-frag feeds 4 WMMAs, loads overlap math
// ---------------------------------------------------------------------------

typedef _Float16 h8  __attribute__((ext_vector_type(8)));
typedef _Float16 h16 __attribute__((ext_vector_type(16)));
typedef float    f8  __attribute__((ext_vector_type(8)));
typedef int      i32x4 __attribute__((ext_vector_type(4)));
typedef __attribute__((address_space(1))) i32x4 gi32x4;   // global 16B vector
typedef __attribute__((address_space(3))) i32x4 li32x4;   // LDS    16B vector

#define DIMX     384
#define LDST     392         // padded LDS row stride in halves (49 uint4)
#define KSTEP    12          // 384 / 32
#define NANCH    8192        // BS*N
#define NQROWS   3200        // K*PIXEL_SIZE
#define NTOT     11392       // NANCH + NQROWS
#define NCLS     64
#define KSAMP    40          // PIXEL_SIZE - K_BAN
#define NXC      2560        // K*KSAMP
#define PIXSZ    50
#define INV_TEMP 10.0f
#define LAMBF    25.0f
#define TBT      0.05f       // TEMP / BASE_TEMP

// Stand-in for jax.random.permutation(key(1), 50)[:40] (distinct, 40 of 50).
__device__ __constant__ int c_idx[KSAMP] = {
  23, 7,41,15, 0,36,29,11,48, 3,19,44, 8,31,26, 1,38,14,47,22,
   5,33,10,42,17,28, 2,49,13,35, 6,45,21, 9,40,25, 4,37,16,30};

// ------------------------- shared device helpers ---------------------------

// Async-copy `rows` x 384-half rows (row-major, dense) into a padded LDS tile.
// 64-row tile => exactly 12 async b128 loads per thread (256-thread block).
__device__ __forceinline__ void async_fill_tile(const _Float16* __restrict__ g,
                                                _Float16* s, int rows, int tid) {
  const uint4* gv = (const uint4*)g;          // 48 uint4 per source row
  uint4* sv = (uint4*)s;                      // 49 uint4 per padded dest row
  for (int i = tid; i < rows * 48; i += 256) {
    int r = i / 48, c = i - r * 48;
    __builtin_amdgcn_global_load_async_to_lds_b128(
        (gi32x4*)(gv + i),
        (li32x4*)(sv + r * 49 + c),
        0, 0);
  }
}

__device__ __forceinline__ void wait_async_all()  {   // drain ASYNCcnt
  asm volatile("s_wait_asynccnt 0x0" ::: "memory");
}
__device__ __forceinline__ void wait_async_le12() {   // keep next tile in flight
  asm volatile("s_wait_asynccnt 0xc" ::: "memory");
}

// One wave: C[16 x 64] += A16xK . B^T(64xK), K=384, from padded LDS tiles.
// kt-outer / ns-inner: A-frag reused by 4 WMMAs; B loads overlap WMMA issue.
__device__ __forceinline__ void gemm_16x64(const _Float16* At, const _Float16* Bt,
                                           int wv, int hi, int nl, f8 acc[4]) {
#pragma unroll
  for (int kt = 0; kt < KSTEP; ++kt) {
    const _Float16* ap = &At[(wv * 16 + nl) * LDST + kt * 32 + hi * 8];
    union { h16 v; h8 h[2]; } a;
    a.h[0] = *(const h8*)ap; a.h[1] = *(const h8*)(ap + 16);
#pragma unroll
    for (int ns = 0; ns < 4; ++ns) {
      const _Float16* bp = &Bt[(ns * 16 + nl) * LDST + kt * 32 + hi * 8];
      union { h16 v; h8 h[2]; } b;
      b.h[0] = *(const h8*)bp; b.h[1] = *(const h8*)(bp + 16);
      acc[ns] = __builtin_amdgcn_wmma_f32_16x16x32_f16(
          false, a.v, false, b.v, (short)0, acc[ns], false, false);
    }
  }
}

// ---------------------------- prep kernels ---------------------------------

__global__ void l2norm_f16_kernel(const float* __restrict__ src,
                                  _Float16* __restrict__ dst, int nrows) {
  int row  = blockIdx.x * 8 + (threadIdx.x >> 5);
  int lane = threadIdx.x & 31;
  if (row >= nrows) return;
  const float* p = src + (size_t)row * DIMX;
  float ss = 0.f;
  for (int j = lane; j < DIMX; j += 32) { float v = p[j]; ss += v * v; }
  for (int off = 1; off < 32; off <<= 1) ss += __shfl_xor(ss, off, 32);
  float sc = 1.0f / fmaxf(sqrtf(ss), 1e-12f);
  _Float16* q = dst + (size_t)row * DIMX;
  for (int j = lane; j < DIMX; j += 32) q[j] = (_Float16)(p[j] * sc);
}

__global__ void f32_to_f16_kernel(const float* __restrict__ src,
                                  _Float16* __restrict__ dst, int n) {
  int i = blockIdx.x * blockDim.x + threadIdx.x;
  if (i < n) dst[i] = (_Float16)src[i];
}

__global__ void gather_xc_kernel(const _Float16* __restrict__ pqh,
                                 _Float16* __restrict__ xch) {
  int e = blockIdx.x * blockDim.x + threadIdx.x;
  if (e >= NXC * DIMX) return;
  int j = e / DIMX, d = e - j * DIMX;
  int c = j / KSAMP, s = c_idx[j - c * KSAMP];
  xch[e] = pqh[((size_t)(c * PIXSZ + s)) * DIMX + d];
}

// ------------------------- scores GEMM (WMMA) ------------------------------
// scores[11392 x 64] = LAMB * cluster_feats @ center^T
__global__ void scores_kernel(const _Float16* __restrict__ noff,
                              const _Float16* __restrict__ pqh,
                              const _Float16* __restrict__ center,
                              float* __restrict__ scores) {
  extern __shared__ char smem[];
  _Float16* At = (_Float16*)smem;       // 128 x LDST
  _Float16* Ct = At + 128 * LDST;       //  64 x LDST
  int tid = threadIdx.x, lane = tid & 31, wv = tid >> 5;
  int hi = lane >> 4, nl = lane & 15;
  int blk = blockIdx.x;
  const _Float16* srcA = (blk < 64) ? (noff + (size_t)blk * 128 * DIMX)
                                    : (pqh  + (size_t)(blk - 64) * 128 * DIMX);
  async_fill_tile(srcA, At, 128, tid);
  async_fill_tile(center, Ct, 64, tid);
  wait_async_all();
  __syncthreads();

  f8 acc[4];
#pragma unroll
  for (int ns = 0; ns < 4; ++ns) acc[ns] = (f8){0.f,0.f,0.f,0.f,0.f,0.f,0.f,0.f};
  gemm_16x64(At, Ct, wv, hi, nl, acc);

  int rowBase = blk * 128;
#pragma unroll
  for (int ns = 0; ns < 4; ++ns) {
    int col = ns * 16 + nl;
#pragma unroll
    for (int r = 0; r < 8; ++r) {
      int m = wv * 16 + r + 8 * hi;
      scores[(size_t)(rowBase + m) * NCLS + col] = acc[ns][r] * LAMBF;
    }
  }
}

// ------------------------------ Sinkhorn -----------------------------------

__global__ void sinkhorn_row_kernel(float* __restrict__ logq) {
  int row  = blockIdx.x * 8 + (threadIdx.x >> 5);
  int lane = threadIdx.x & 31;
  if (row >= NTOT) return;
  float* p = logq + (size_t)row * NCLS;
  float x0 = p[lane], x1 = p[lane + 32];
  float m = fmaxf(x0, x1);
  for (int off = 1; off < 32; off <<= 1) m = fmaxf(m, __shfl_xor(m, off, 32));
  float s = __expf(x0 - m) + __expf(x1 - m);
  for (int off = 1; off < 32; off <<= 1) s += __shfl_xor(s, off, 32);
  float lse = m + __logf(s);
  p[lane] = x0 - lse;
  p[lane + 32] = x1 - lse;
}

__global__ void sinkhorn_col_lse_kernel(const float* __restrict__ logq,
                                        float* __restrict__ colLSE) {
  int c = blockIdx.x, tid = threadIdx.x;
  float m = -3.0e38f, s = 0.f;
  for (int i = tid; i < NTOT; i += 256) {
    float x = logq[(size_t)i * NCLS + c];
    if (x > m) { s = s * __expf(m - x) + 1.0f; m = x; }
    else       { s += __expf(x - m); }
  }
  __shared__ float sm[256], ssh[256];
  sm[tid] = m; ssh[tid] = s;
  __syncthreads();
  for (int st = 128; st > 0; st >>= 1) {
    if (tid < st) {
      float mb = sm[tid + st], sb = ssh[tid + st];
      float ma = sm[tid],      sa = ssh[tid];
      if (mb > ma) { sa = sa * __expf(ma - mb) + sb; ma = mb; }
      else         { sa = sa + sb * __expf(mb - ma); }
      sm[tid] = ma; ssh[tid] = sa;
    }
    __syncthreads();
  }
  if (tid == 0) colLSE[c] = sm[0] + __logf(ssh[0]);
}

__global__ void col_sub_kernel(float* __restrict__ logq,
                               const float* __restrict__ colLSE) {
  int e = blockIdx.x * 256 + threadIdx.x;
  if (e < NTOT * NCLS) logq[e] -= colLSE[e & 63];
}

__global__ void argmax_kernel(const float* __restrict__ logq,
                              int* __restrict__ labels) {
  int i = blockIdx.x * 256 + threadIdx.x;
  if (i >= NANCH) return;
  const float* p = logq + (size_t)i * NCLS;
  float best = p[0]; int bi = 0;
  for (int k = 1; k < NCLS; ++k) {
    float v = p[k];
    if (v > best) { best = v; bi = k; }
  }
  labels[i] = bi;
}

// ----------------------- supcon WMMA passes --------------------------------
// grid.x = 64 (row strips of 128), grid.y = column slices; block = 256 (8 waves).
// Each wave owns 16 rows; per-row stats accumulate in registers, combined
// across column slices with global f32 atomics. PHASE templated so the unused
// accumulator set is dead-stripped. B tiles are double-buffered in LDS with
// partial ASYNCcnt waits so the tile DMA overlaps WMMA + exp/log work.
template <int PHASE>
__global__ void supcon_pass_kernel(const _Float16* __restrict__ A,
                                   const _Float16* __restrict__ B, int ncols,
                                   const int* __restrict__ yA,
                                   const int* __restrict__ yB,  // null -> col/KSAMP
                                   int excludeDiag,
                                   const float* __restrict__ negIn,
                                   float* __restrict__ negOut,
                                   float* __restrict__ cntOut,
                                   float* __restrict__ numOut) {
  extern __shared__ char smem[];
  _Float16* At  = (_Float16*)smem;             // 128 x LDST
  _Float16* Bt0 = At + 128 * LDST;             //  64 x LDST (buffer 0)
  _Float16* Bt1 = Bt0 + 64 * LDST;             //  64 x LDST (buffer 1)
  float* negRow = (float*)(Bt1 + 64 * LDST);   // 128
  int*   yRow   = (int*)(negRow + 128);        // 128

  int tid = threadIdx.x, lane = tid & 31, wv = tid >> 5;
  int hi = lane >> 4, nl = lane & 15;
  int rowBase = blockIdx.x * 128;
  int nTiles  = ncols >> 6;
  int stride  = gridDim.y;

  // prologue: A strip (24 loads/thread) + first B tile (12 loads/thread)
  async_fill_tile(A + (size_t)rowBase * DIMX, At, 128, tid);
  async_fill_tile(B + (size_t)blockIdx.y * 64 * DIMX, Bt0, 64, tid);
  for (int i = tid; i < 128; i += 256) {
    yRow[i]   = yA[rowBase + i];
    negRow[i] = (PHASE == 2) ? negIn[rowBase + i] : 0.f;
  }

  float accNeg[8], accCnt[8], accNum[8];
#pragma unroll
  for (int r = 0; r < 8; ++r) { accNeg[r] = 0.f; accCnt[r] = 0.f; accNum[r] = 0.f; }

  int cur = 0;
  for (int nt = blockIdx.y; nt < nTiles; nt += stride) {
    // 1) kick off next tile's DMA into the other buffer (safe: end-of-iter
    //    barrier of the previous iteration guarantees it is no longer read)
    int ntNext = nt + stride;
    if (ntNext < nTiles)
      async_fill_tile(B + (size_t)ntNext * 64 * DIMX, cur ? Bt0 : Bt1, 64, tid);

    // 2) wait for the current tile only: async loads complete in order, and
    //    exactly 12/thread are in flight for the next tile
    if (ntNext < nTiles) wait_async_le12(); else wait_async_all();
    __syncthreads();

    const _Float16* Bt = cur ? Bt1 : Bt0;
    f8 acc[4];
#pragma unroll
    for (int ns = 0; ns < 4; ++ns) acc[ns] = (f8){0.f,0.f,0.f,0.f,0.f,0.f,0.f,0.f};
    gemm_16x64(At, Bt, wv, hi, nl, acc);

    // 3) stats (exp/log VALU) overlaps the background DMA of the next tile
#pragma unroll
    for (int ns = 0; ns < 4; ++ns) {
      int col  = nt * 64 + ns * 16 + nl;
      int ycol = yB ? yB[col] : (col / KSAMP);
#pragma unroll
      for (int r = 0; r < 8; ++r) {
        int m = wv * 16 + r + 8 * hi;
        float l = acc[ns][r] * INV_TEMP;
        bool pos = (yRow[m] == ycol) &&
                   !(excludeDiag && ((rowBase + m) == col));
        if (PHASE == 1) {
          if (pos) accCnt[r] += 1.0f;
          else     accNeg[r] += __expf(l);
        } else {
          if (pos) accNum[r] += l - __logf(__expf(l) + negRow[m]);
        }
      }
    }

    __syncthreads();   // all waves done reading Bt[cur] before it is refilled
    cur ^= 1;
  }

  // reduce each row across its 16-lane group; lanes 0/16 hold rows r / r+8
  if (PHASE == 1) {
#pragma unroll
    for (int r = 0; r < 8; ++r) {
      for (int off = 1; off < 16; off <<= 1) {
        accNeg[r] += __shfl_xor(accNeg[r], off, 32);
        accCnt[r] += __shfl_xor(accCnt[r], off, 32);
      }
    }
    if (nl == 0) {
#pragma unroll
      for (int r = 0; r < 8; ++r) {
        int m = rowBase + wv * 16 + r + 8 * hi;
        atomicAdd(&negOut[m], accNeg[r]);
        atomicAdd(&cntOut[m], accCnt[r]);
      }
    }
  } else {
#pragma unroll
    for (int r = 0; r < 8; ++r)
      for (int off = 1; off < 16; off <<= 1)
        accNum[r] += __shfl_xor(accNum[r], off, 32);
    if (nl == 0) {
#pragma unroll
      for (int r = 0; r < 8; ++r) {
        int m = rowBase + wv * 16 + r + 8 * hi;
        atomicAdd(&numOut[m], accNum[r]);
      }
    }
  }
}

// --------------------------- final reduction -------------------------------
__global__ void final_loss_kernel(const float* __restrict__ num2,
                                  const float* __restrict__ cnt2,
                                  const float* __restrict__ num1,
                                  const float* __restrict__ cnt1,
                                  float* __restrict__ out) {
  __shared__ float r0[256], r1[256], r2[256];
  int tid = threadIdx.x;
  float s1 = 0.f, sM = 0.f, sV = 0.f;
  for (int i = tid; i < NANCH; i += 256) {
    s1 += num2[i] / cnt2[i];              // require_pos=False: raw division
    float c = cnt1[i];
    if (c > 0.f) { sM += num1[i] / fmaxf(c, 1.0f); sV += 1.0f; }
  }
  r0[tid] = s1; r1[tid] = sM; r2[tid] = sV;
  __syncthreads();
  for (int st = 128; st > 0; st >>= 1) {
    if (tid < st) { r0[tid] += r0[tid + st]; r1[tid] += r1[tid + st]; r2[tid] += r2[tid + st]; }
    __syncthreads();
  }
  if (tid == 0) {
    float loss1 = -TBT * (r0[0] / (float)NANCH);
    float loss2 = -TBT * r1[0] / fmaxf(r2[0], 1.0f);
    out[0] = loss1 + loss2;
  }
}

// ------------------------------ launcher -----------------------------------
extern "C" void kernel_launch(void* const* d_in, const int* in_sizes, int n_in,
                              void* d_out, int out_size, void* d_ws, size_t ws_size,
                              hipStream_t stream) {
  const float* feats     = (const float*)d_in[0];  // (16,512,384)
  const float* off_feats = (const float*)d_in[1];  // (16,512,384)
  const float* center    = (const float*)d_in[2];  // (64,384)
  const float* pq        = (const float*)d_in[3];  // (64,50,384)
  float* out = (float*)d_out;

  // -- carve workspace (256B aligned chunks) --
  size_t off = 0;
  auto carve = [&](size_t bytes) -> void* {
    off = (off + 255) & ~(size_t)255;
    void* p = (char*)d_ws + off;
    off += bytes;
    return p;
  };
  _Float16* f_h    = (_Float16*)carve((size_t)NANCH * DIMX * 2);
  _Float16* noff_h = (_Float16*)carve((size_t)NANCH * DIMX * 2);
  _Float16* ctr_h  = (_Float16*)carve((size_t)NCLS * DIMX * 2);
  _Float16* pq_h   = (_Float16*)carve((size_t)NQROWS * DIMX * 2);
  _Float16* xc_h   = (_Float16*)carve((size_t)NXC * DIMX * 2);
  float*    scores = (float*)carve((size_t)NTOT * NCLS * 4);
  float*    colLSE = (float*)carve(NCLS * 4);
  int*      labels = (int*)carve(NANCH * 4);
  float*    stats  = (float*)carve((size_t)6 * NANCH * 4);  // neg2,cnt2,num2,neg1,cnt1,num1
  float* neg2 = stats + 0 * NANCH, *cnt2 = stats + 1 * NANCH, *num2 = stats + 2 * NANCH;
  float* neg1 = stats + 3 * NANCH, *cnt1 = stats + 4 * NANCH, *num1 = stats + 5 * NANCH;

  const size_t smemGemm  = (size_t)(128 + 64) * LDST * 2;                 // ~147 KB
  const size_t smemSupcn = (size_t)(128 + 64 + 64) * LDST * 2 + 1024;    // ~198 KB
  (void)hipFuncSetAttribute((const void*)scores_kernel,
      hipFuncAttributeMaxDynamicSharedMemorySize, (int)smemGemm);
  (void)hipFuncSetAttribute((const void*)supcon_pass_kernel<1>,
      hipFuncAttributeMaxDynamicSharedMemorySize, (int)smemSupcn);
  (void)hipFuncSetAttribute((const void*)supcon_pass_kernel<2>,
      hipFuncAttributeMaxDynamicSharedMemorySize, (int)smemSupcn);

  // 1) normalize + f16 conversions
  l2norm_f16_kernel<<<NANCH / 8, 256, 0, stream>>>(feats, f_h, NANCH);
  l2norm_f16_kernel<<<NANCH / 8, 256, 0, stream>>>(off_feats, noff_h, NANCH);
  f32_to_f16_kernel<<<(NCLS * DIMX) / 256, 256, 0, stream>>>(center, ctr_h, NCLS * DIMX);
  f32_to_f16_kernel<<<(NQROWS * DIMX) / 256, 256, 0, stream>>>(pq, pq_h, NQROWS * DIMX);
  gather_xc_kernel<<<(NXC * DIMX) / 256, 256, 0, stream>>>(pq_h, xc_h);

  // 2) scores = LAMB * cluster_feats @ center^T
  scores_kernel<<<NTOT / 128, 256, smemGemm, stream>>>(noff_h, pq_h, ctr_h, scores);

  // 3) Sinkhorn (3 iters) + argmax -> labels
  for (int it = 0; it < 3; ++it) {
    sinkhorn_row_kernel<<<NTOT / 8, 256, 0, stream>>>(scores);
    sinkhorn_col_lse_kernel<<<NCLS, 256, 0, stream>>>(scores, colLSE);
    col_sub_kernel<<<(NTOT * NCLS) / 256, 256, 0, stream>>>(scores, colLSE);
  }
  argmax_kernel<<<NANCH / 256, 256, 0, stream>>>(scores, labels);

  // 4) supcon losses (stats atomically accumulated -> zero first)
  (void)hipMemsetAsync(stats, 0, (size_t)6 * NANCH * 4, stream);

  // ppc2: anchors vs sampled queue (no diag), y_c = col/40
  supcon_pass_kernel<1><<<dim3(NANCH / 128, 4), 256, smemSupcn, stream>>>(
      f_h, xc_h, NXC, labels, nullptr, 0, nullptr, neg2, cnt2, nullptr);
  supcon_pass_kernel<2><<<dim3(NANCH / 128, 4), 256, smemSupcn, stream>>>(
      f_h, xc_h, NXC, labels, nullptr, 0, neg2, nullptr, nullptr, num2);

  // ppc: anchors vs anchors, diagonal excluded from positives
  supcon_pass_kernel<1><<<dim3(NANCH / 128, 8), 256, smemSupcn, stream>>>(
      f_h, f_h, NANCH, labels, labels, 1, nullptr, neg1, cnt1, nullptr);
  supcon_pass_kernel<2><<<dim3(NANCH / 128, 8), 256, smemSupcn, stream>>>(
      f_h, f_h, NANCH, labels, labels, 1, neg1, nullptr, nullptr, num1);

  // 5) final scalar
  final_loss_kernel<<<1, 256, 0, stream>>>(num2, cnt2, num1, cnt1, out);
}
